// MelBandSplit_78872779424083
// MI455X (gfx1250) — compile-verified, matrix operand order checked by reading, not run on hardware
//
#include <hip/hip_runtime.h>
#include <hip/hip_bf16.h>

typedef __attribute__((ext_vector_type(16))) __bf16 v16bf;
typedef __attribute__((ext_vector_type(8)))  __bf16 v8bf;
typedef __attribute__((ext_vector_type(8)))  float  v8f;

#define NBANDS 61
#define FEAT   512
#define MROWS  2048          // b*t = 4*512
#define KSTRIDE 256          // max padded K (d<=256)
#define TOTAL_KSTEPS 129     // sum over bands of ceil(d/32)
#define PACK_ELEMS ((size_t)TOTAL_KSTEPS * 32 /*ntiles*/ * 32 /*lanes*/ * 16 /*elems*/)

struct Ptrs {
    const float* W[NBANDS];
    const float* g[NBANDS];
    const float* b[NBANDS];
};

struct BandInfo { int lo; int d; int ksteps; int cumk; };

__device__ __forceinline__ BandInfo band_info(int i) {
    // widths: 32 bands of 8, 16 of 16, 8 of 32, 4 of 64, 1 of 1 (freq bins); d = width*4
    BandInfo bi;
    if (i < 32)      { bi.lo = i * 8;              bi.d = 32;  bi.ksteps = 1; bi.cumk = i; }
    else if (i < 48) { bi.lo = 256 + (i - 32) * 16; bi.d = 64;  bi.ksteps = 2; bi.cumk = 32 + (i - 32) * 2; }
    else if (i < 56) { bi.lo = 512 + (i - 48) * 32; bi.d = 128; bi.ksteps = 4; bi.cumk = 64 + (i - 48) * 4; }
    else if (i < 60) { bi.lo = 768 + (i - 56) * 64; bi.d = 256; bi.ksteps = 8; bi.cumk = 96 + (i - 56) * 8; }
    else             { bi.lo = 1024;               bi.d = 4;   bi.ksteps = 1; bi.cumk = 128; }
    return bi;
}

__device__ __forceinline__ unsigned bf16_rne(float f) {
    unsigned u = __float_as_uint(f);
    return (u + 0x7FFFu + ((u >> 16) & 1u)) >> 16;
}

// -------------------------------------------------------------------------
// Pack W*gamma into bf16 hi/lo pair in exact WMMA-B-fragment lane order:
// slot s (within band) = ((ntile*ksteps + kstep)*32 + lane), 16 bf16 per slot.
// Element e of lane L covers K = kstep*32 + (e<8 ? e : e+8) + ((L>=16)?8:0),
// N = ntile*16 + (L&15)  -- mirrors the 16-bit A-matrix VGPR striping.
// -------------------------------------------------------------------------
__global__ __launch_bounds__(256) void pack_w_kernel(Ptrs p,
                                                     unsigned short* __restrict__ hi,
                                                     unsigned short* __restrict__ lo) {
    int s = blockIdx.x * 256 + threadIdx.x;
    if (s >= TOTAL_KSTEPS * 1024) return;

    int band = 0, acc = 0;
    for (int i = 0; i < NBANDS; ++i) {
        BandInfo bi = band_info(i);
        int cnt = bi.ksteps * 1024;           // 32 ntiles * 32 lanes per kstep
        if (s < acc + cnt) { band = i; break; }
        acc += cnt;
    }
    BandInfo bi = band_info(band);
    int ls    = s - acc;
    int lane  = ls & 31;
    int t     = ls >> 5;
    int kstep = t % bi.ksteps;
    int ntile = t / bi.ksteps;
    int n     = ntile * 16 + (lane & 15);
    int kbase = kstep * 32 + ((lane >> 4) & 1) * 8;

    const float* W = p.W[band];
    const float* g = p.g[band];
    size_t ob = (size_t)bi.cumk * 16384 + (size_t)ls * 16;

#pragma unroll
    for (int e = 0; e < 16; ++e) {
        int k = kbase + (e < 8 ? e : e + 8);
        float w = 0.f;
        if (k < bi.d) w = W[n * bi.d + k] * g[k];
        unsigned h = bf16_rne(w);
        float rem  = w - __uint_as_float(h << 16);
        unsigned l = bf16_rne(rem);
        hi[ob + e] = (unsigned short)h;
        lo[ob + e] = (unsigned short)l;
    }
}

// -------------------------------------------------------------------------
// Main kernel: block = (band, 16-row M tile). 8 waves, each owns 4 N-tiles.
// bf16x3 emulation of f32 GEMM on v_wmma_f32_16x16x32_bf16.
// -------------------------------------------------------------------------
__global__ __launch_bounds__(256) void melband_main_kernel(const float* __restrict__ x,
                                                           const unsigned short* __restrict__ whi,
                                                           const unsigned short* __restrict__ wlo,
                                                           Ptrs p,
                                                           float* __restrict__ out) {
    __shared__ __align__(16) float          As[16 * KSTRIDE];
    __shared__ __align__(32) unsigned short Ah[16 * KSTRIDE];
    __shared__ __align__(32) unsigned short Al[16 * KSTRIDE];
    __shared__ float rowScale[16];

    int band  = blockIdx.x / 128;
    int mtile = blockIdx.x % 128;
    BandInfo bi = band_info(band);
    int Kpad = bi.ksteps * 32;

    int tid = threadIdx.x;
    int r   = tid >> 4;     // staging row 0..15
    int kc  = tid & 15;     // staging k phase

    // Stage segment: seg[row m, col k] = x[b, c=k&3, t, f=lo + k/4]
    {
        int m  = mtile * 16 + r;
        int b_ = m >> 9;
        int t_ = m & 511;
        const size_t CSTR = (size_t)512 * 1025;
        size_t base_bt = (size_t)b_ * 4 * CSTR + (size_t)t_ * 1025;
        for (int k = kc; k < Kpad; k += 16) {
            float v = 0.f;
            if (k < bi.d) {
                int f = bi.lo + (k >> 2);
                int c = k & 3;
                v = x[base_bt + (size_t)c * CSTR + f];
            }
            As[r * KSTRIDE + k] = v;
        }
    }
    __syncthreads();

    // Per-row scale = sqrt(d) / max(||row||, 1e-12)
    if (tid < 16) {
        float s = 0.f;
        for (int k = 0; k < bi.d; ++k) {
            float v = As[tid * KSTRIDE + k];
            s += v * v;
        }
        rowScale[tid] = sqrtf((float)bi.d) / fmaxf(sqrtf(s), 1e-12f);
    }
    __syncthreads();

    // Scale + split into bf16 hi/lo
    for (int k = kc; k < Kpad; k += 16) {
        float a = As[r * KSTRIDE + k] * rowScale[r];
        unsigned h = bf16_rne(a);
        float rem  = a - __uint_as_float(h << 16);
        unsigned l = bf16_rne(rem);
        Ah[r * KSTRIDE + k] = (unsigned short)h;
        Al[r * KSTRIDE + k] = (unsigned short)l;
    }
    __syncthreads();

    int wave = tid >> 5;
    int lane = tid & 31;
    int arow  = lane & 15;                 // A-fragment: lanes 0-15 and 16-31 both hold rows 0-15
    int khalf = ((lane >> 4) & 1) * 8;     // upper half-wave covers K+8

    v8f acc[4] = {};
    size_t wband = (size_t)bi.cumk * 16384;

    for (int kstep = 0; kstep < bi.ksteps; ++kstep) {
        int ko = kstep * 32 + khalf;
        v16bf ah, al;
        ((v8bf*)&ah)[0] = *(const v8bf*)&Ah[arow * KSTRIDE + ko];
        ((v8bf*)&ah)[1] = *(const v8bf*)&Ah[arow * KSTRIDE + ko + 16];
        ((v8bf*)&al)[0] = *(const v8bf*)&Al[arow * KSTRIDE + ko];
        ((v8bf*)&al)[1] = *(const v8bf*)&Al[arow * KSTRIDE + ko + 16];

#pragma unroll
        for (int jj = 0; jj < 4; ++jj) {
            int ntile = wave * 4 + jj;
            size_t fb = wband + ((size_t)(ntile * bi.ksteps + kstep) * 32 + lane) * 16;
            v16bf bh = *(const v16bf*)(whi + fb);
            v16bf bl = *(const v16bf*)(wlo + fb);
            acc[jj] = __builtin_amdgcn_wmma_f32_16x16x32_bf16(false, ah, false, bh, (short)0, acc[jj], false, false);
            acc[jj] = __builtin_amdgcn_wmma_f32_16x16x32_bf16(false, ah, false, bl, (short)0, acc[jj], false, false);
            acc[jj] = __builtin_amdgcn_wmma_f32_16x16x32_bf16(false, al, false, bh, (short)0, acc[jj], false, false);
        }
    }

    // Epilogue: C/D layout -> lane owns fixed N = lane&15; VGPR r is row M=r (+8 for upper half-wave)
    int mrow_off = ((lane >> 4) & 1) * 8;
#pragma unroll
    for (int jj = 0; jj < 4; ++jj) {
        int n = (wave * 4 + jj) * 16 + (lane & 15);
        float bias = p.b[band][n];
#pragma unroll
        for (int rr = 0; rr < 8; ++rr) {
            int mm = mtile * 16 + rr + mrow_off;
            out[((size_t)mm * NBANDS + band) * FEAT + n] = acc[jj][rr] + bias;
        }
    }
}

extern "C" void kernel_launch(void* const* d_in, const int* in_sizes, int n_in,
                              void* d_out, int out_size, void* d_ws, size_t ws_size,
                              hipStream_t stream) {
    const float* x = (const float*)d_in[0];
    Ptrs p;
    for (int i = 0; i < NBANDS; ++i) {
        p.g[i] = (const float*)d_in[1 + i];
        p.W[i] = (const float*)d_in[1 + NBANDS + i];
        p.b[i] = (const float*)d_in[1 + 2 * NBANDS + i];
    }
    unsigned short* whi = (unsigned short*)d_ws;
    unsigned short* wlo = whi + PACK_ELEMS;
    float* out = (float*)d_out;

    int pack_threads = TOTAL_KSTEPS * 1024;
    pack_w_kernel<<<(pack_threads + 255) / 256, 256, 0, stream>>>(p, whi, wlo);
    melband_main_kernel<<<NBANDS * 128, 256, 0, stream>>>(x, whi, wlo, p, out);
}
